// TopKRouter_19533511262529
// MI455X (gfx1250) — matrix operand order repaired and checked by reading, not compiled, not run on hardware
//
#include <hip/hip_runtime.h>
#include <hip/hip_bf16.h>
#include <math.h>

// ---------------- problem constants (from reference) ----------------
#define NUM_EXPERTS 64
#define TOP_K       8
#define D_DIM       2048
#define TOKENS      (4 * 4096)        // B*S = 16384
#define TILE_M      16                // tokens per wave tile (WMMA M)
#define WAVES_PER_BLOCK 8
#define BLOCK_THREADS   (WAVES_PER_BLOCK * 32)
#define K_CHUNK     128               // gate_w K-slab staged in LDS per block
#define CHUNKS      (D_DIM / K_CHUNK) // 16

// flat output offsets (tuple concatenated in return order)
#define OFF_IDX    0
#define OFF_W      (TOKENS * TOP_K)                  // 131072
#define OFF_ZLOSS  (OFF_W + TOKENS * TOP_K)          // 262144
#define OFF_COUNTS (OFF_ZLOSS + 1)                   // 262145
#define OFF_TOTAL  (OFF_COUNTS + NUM_EXPERTS)        // 262209
#define OFF_BIAS   (OFF_TOTAL + 1)                   // 262210
#define OFF_UTIL   (OFF_BIAS + NUM_EXPERTS)          // 262274

typedef __bf16 bf16_t;
typedef bf16_t v16bf __attribute__((ext_vector_type(16)));
typedef float  v8f   __attribute__((ext_vector_type(8)));

__device__ __forceinline__ void split_bf16(float f, bf16_t& hi, bf16_t& lo) {
    hi = (bf16_t)f;                    // RNE convert
    lo = (bf16_t)(f - (float)hi);      // residual -> ~fp32 accuracy with 3 products
}

// ---------------------------------------------------------------------------
// Kernel 1: gate GEMM (bf16 hi/lo split on v_wmma_f32_16x16x32_bf16) with
// gate_w staged+pre-split in LDS per block, then per-token top-8 select,
// softmax, z-loss partial and expert-count histogram (block-level LDS
// reduction, one global flush per block).
// One wave = one 16-token tile x all 64 experts (4 WMMA N-tiles).
// ---------------------------------------------------------------------------
__global__ __launch_bounds__(BLOCK_THREADS)
void router_gemm_topk_kernel(const float* __restrict__ x,
                             const float* __restrict__ gate_w,
                             const float* __restrict__ expert_bias,
                             float* __restrict__ out_idx,
                             float* __restrict__ out_w,
                             float* __restrict__ ws_counts,   // [64]
                             float* __restrict__ ws_zacc)     // [1]
{
    __shared__ float  s_logits[WAVES_PER_BLOCK][TILE_M * NUM_EXPERTS]; // 32 KB
    __shared__ __align__(32) bf16_t s_bhi[NUM_EXPERTS * K_CHUNK];      // 16 KB
    __shared__ __align__(32) bf16_t s_blo[NUM_EXPERTS * K_CHUNK];      // 16 KB
    __shared__ float  s_bias[NUM_EXPERTS];
    __shared__ float  s_hist[NUM_EXPERTS];   // block-local expert histogram
    __shared__ float  s_zsum[1];             // block-local sum of logz^2

    const int lane = threadIdx.x & 31;
    const int wave = threadIdx.x >> 5;

    if (threadIdx.x < NUM_EXPERTS) {
        s_bias[threadIdx.x] = expert_bias[threadIdx.x];
        s_hist[threadIdx.x] = 0.0f;
    }
    if (threadIdx.x == 0) s_zsum[0] = 0.0f;

    const int tile   = blockIdx.x * WAVES_PER_BLOCK + wave;
    const int token0 = tile * TILE_M;

    const int n = lane & 15;   // A row (M) and B column (N) handled by this lane
    const int h = lane >> 4;   // half-wave selector

    const float* xrow = x + (size_t)(token0 + n) * D_DIM;

    v8f acc0 = {}, acc1 = {}, acc2 = {}, acc3 = {};

    for (int c = 0; c < CHUNKS; ++c) {
        const int kbase = c * K_CHUNK;

        // ---- cooperatively stage gate_w[0:64][kbase:kbase+128] split to hi/lo
        __syncthreads();   // previous chunk's readers are done
        for (int v = threadIdx.x; v < (NUM_EXPERTS * K_CHUNK) / 4; v += BLOCK_THREADS) {
            const int e  = (v * 4) / K_CHUNK;
            const int kc = (v * 4) % K_CHUNK;
            const float4 f = *(const float4*)(gate_w + (size_t)e * D_DIM + kbase + kc);
            bf16_t hi, lo;
            const int base = e * K_CHUNK + kc;
            split_bf16(f.x, hi, lo); s_bhi[base + 0] = hi; s_blo[base + 0] = lo;
            split_bf16(f.y, hi, lo); s_bhi[base + 1] = hi; s_blo[base + 1] = lo;
            split_bf16(f.z, hi, lo); s_bhi[base + 2] = hi; s_blo[base + 2] = lo;
            split_bf16(f.w, hi, lo); s_bhi[base + 3] = hi; s_blo[base + 3] = lo;
        }
        __syncthreads();

        // ---- 4 WMMA K-steps over the staged chunk
        for (int kk = 0; kk < K_CHUNK; kk += 32) {
            const int k0 = kbase + kk;
            // speculative prefetch of next x K-slab (global_prefetch_b8)
            if (k0 + 32 < D_DIM)
                __builtin_prefetch(xrow + k0 + 32, 0, 1);

            // A fragment (16x32 bf16), contiguous k order, same as B:
            //   halfs[0..7]  = x[m][k0 + h*16 + 0..7]
            //   halfs[8..15] = x[m][k0 + h*16 + 8 + 0..7]
            float af[16];
            {
                const float4* p0 = (const float4*)(xrow + k0 + h * 16);
                const float4* p1 = (const float4*)(xrow + k0 + h * 16 + 8);
                float4 a0 = p0[0], a1 = p0[1], a2 = p1[0], a3 = p1[1];
                af[0]=a0.x;  af[1]=a0.y;  af[2]=a0.z;  af[3]=a0.w;
                af[4]=a1.x;  af[5]=a1.y;  af[6]=a1.z;  af[7]=a1.w;
                af[8]=a2.x;  af[9]=a2.y;  af[10]=a2.z; af[11]=a2.w;
                af[12]=a3.x; af[13]=a3.y; af[14]=a3.z; af[15]=a3.w;
            }
            v16bf a_hi, a_lo;
#pragma unroll
            for (int i = 0; i < 16; ++i) {
                bf16_t hi, lo; split_bf16(af[i], hi, lo);
                a_hi[i] = hi; a_lo[i] = lo;
            }

            // B fragments (32x16 bf16) straight from LDS, contiguous k order, same as A:
            //   halfs[i] = gate_w[e][k0 + h*16 + i], i = 0..15 (contiguous)
#pragma unroll
            for (int t = 0; t < 4; ++t) {
                const int boff = (t * 16 + n) * K_CHUNK + kk + h * 16;
                const v16bf b_hi = *(const v16bf*)&s_bhi[boff];
                const v16bf b_lo = *(const v16bf*)&s_blo[boff];
                v8f acc = (t == 0) ? acc0 : (t == 1) ? acc1 : (t == 2) ? acc2 : acc3;
                // fp32-accurate product via 3-term bf16 split
                acc = __builtin_amdgcn_wmma_f32_16x16x32_bf16(false, a_hi, false, b_hi,
                                                              (short)0, acc, false, false);
                acc = __builtin_amdgcn_wmma_f32_16x16x32_bf16(false, a_hi, false, b_lo,
                                                              (short)0, acc, false, false);
                acc = __builtin_amdgcn_wmma_f32_16x16x32_bf16(false, a_lo, false, b_hi,
                                                              (short)0, acc, false, false);
                if      (t == 0) acc0 = acc;
                else if (t == 1) acc1 = acc;
                else if (t == 2) acc2 = acc;
                else             acc3 = acc;
            }
        }
    }

    // ---- spill logits to LDS using the documented C/D layout:
    // VGPR r, lanes 0-15: (M=r, N=lane); lanes 16-31: (M=8+r, N=lane-16)
#pragma unroll
    for (int r = 0; r < 8; ++r) {
        const int row = r + 8 * h;
        s_logits[wave][row * NUM_EXPERTS +  0 + n] = acc0[r];
        s_logits[wave][row * NUM_EXPERTS + 16 + n] = acc1[r];
        s_logits[wave][row * NUM_EXPERTS + 32 + n] = acc2[r];
        s_logits[wave][row * NUM_EXPERTS + 48 + n] = acc3[r];
    }

    __syncthreads();

    // ---- per-token routing: lanes 0..15 each own one token of this wave's tile
    if (lane < TILE_M) {
        const int token = token0 + lane;
        const float* L = &s_logits[wave][lane * NUM_EXPERTS];

        // router z-loss on unbiased logits: logsumexp
        float mx = -INFINITY;
#pragma unroll
        for (int e = 0; e < NUM_EXPERTS; ++e) mx = fmaxf(mx, L[e]);
        float s = 0.0f;
#pragma unroll
        for (int e = 0; e < NUM_EXPERTS; ++e) s += expf(L[e] - mx);
        const float logz = mx + logf(s);
        atomicAdd(&s_zsum[0], logz * logz);          // LDS atomic (ds_add_f32)

        // top-8 on biased logits; strict '>' scan == lowest-index tie-break
        unsigned long long taken = 0ull;
        int   idx[TOP_K];
        float sel[TOP_K];
#pragma unroll
        for (int k = 0; k < TOP_K; ++k) {
            float best = -INFINITY;
            int   bi   = 0;
            for (int e = 0; e < NUM_EXPERTS; ++e) {
                if (!((taken >> e) & 1ull)) {
                    const float bl = L[e] + s_bias[e];
                    if (bl > best) { best = bl; bi = e; }
                }
            }
            taken |= (1ull << bi);
            idx[k] = bi;
            sel[k] = L[bi];   // UNBIASED logit for the softmax
        }

        // softmax over the 8 selected unbiased logits
        float m2 = -INFINITY;
#pragma unroll
        for (int k = 0; k < TOP_K; ++k) m2 = fmaxf(m2, sel[k]);
        float sum = 0.0f;
        float w[TOP_K];
#pragma unroll
        for (int k = 0; k < TOP_K; ++k) { w[k] = expf(sel[k] - m2); sum += w[k]; }
        const float inv = 1.0f / sum;

#pragma unroll
        for (int k = 0; k < TOP_K; ++k) {
            out_idx[(size_t)token * TOP_K + k] = (float)idx[k];
            out_w  [(size_t)token * TOP_K + k] = w[k] * inv;
            atomicAdd(&s_hist[idx[k]], 1.0f);        // LDS atomic histogram
        }
    }

    // ---- flush block-local statistics: 64+1 global atomics per block
    __syncthreads();
    if (threadIdx.x < NUM_EXPERTS) {
        const float cnt = s_hist[threadIdx.x];
        if (cnt != 0.0f) atomicAdd(&ws_counts[threadIdx.x], cnt);
    }
    if (threadIdx.x == BLOCK_THREADS - 1)
        atomicAdd(ws_zacc, s_zsum[0]);
}

// ---------------------------------------------------------------------------
// Kernel 2: finalize load-balancing statistics (64 threads, 1 block)
// ---------------------------------------------------------------------------
__global__ void router_finalize_kernel(const float* __restrict__ expert_bias,
                                       const float* __restrict__ expert_counts,
                                       const float* __restrict__ total_tokens,
                                       const float* __restrict__ ws_counts,
                                       const float* __restrict__ ws_zacc,
                                       float* __restrict__ out)
{
    const int e = threadIdx.x;
    const float nc   = expert_counts[e] + ws_counts[e];
    const float nt   = total_tokens[0] + (float)TOKENS;
    const float load = nc / (nt + 1e-8f);
    const float nb   = expert_bias[e] - 0.001f * (load - 1.0f / (float)NUM_EXPERTS);

    out[OFF_COUNTS + e] = nc;
    out[OFF_BIAS   + e] = nb;
    out[OFF_UTIL   + e] = load;
    if (e == 0) {
        out[OFF_ZLOSS] = 1e-5f * ws_zacc[0] / (float)TOKENS;  // coeff * mean(logz^2)
        out[OFF_TOTAL] = nt;
    }
}

// ---------------------------------------------------------------------------
extern "C" void kernel_launch(void* const* d_in, const int* in_sizes, int n_in,
                              void* d_out, int out_size, void* d_ws, size_t ws_size,
                              hipStream_t stream) {
    const float* x             = (const float*)d_in[0];
    const float* gate_w        = (const float*)d_in[1];
    const float* expert_bias   = (const float*)d_in[2];
    const float* expert_counts = (const float*)d_in[3];
    const float* total_tokens  = (const float*)d_in[4];
    float* out = (float*)d_out;
    float* ws  = (float*)d_ws;           // [0..63] counts, [64] z accumulator

    // zero scratch each call (memset node is graph-capture legal)
    hipMemsetAsync(ws, 0, (NUM_EXPERTS + 1) * sizeof(float), stream);

    const int num_tiles  = TOKENS / TILE_M;              // 1024
    const int num_blocks = num_tiles / WAVES_PER_BLOCK;  // 128

    router_gemm_topk_kernel<<<num_blocks, BLOCK_THREADS, 0, stream>>>(
        x, gate_w, expert_bias,
        out + OFF_IDX, out + OFF_W,
        ws, ws + NUM_EXPERTS);

    router_finalize_kernel<<<1, NUM_EXPERTS, 0, stream>>>(
        expert_bias, expert_counts, total_tokens,
        ws, ws + NUM_EXPERTS, out);
}
